// Conv2dSelfAttention_20255065768191
// MI455X (gfx1250) — compile-verified, matrix operand order checked by reading, not run on hardware
//
#include <hip/hip_runtime.h>
#include <hip/hip_bf16.h>
#include <stdint.h>

// ---------------------------------------------------------------------------
// CDNA5 / gfx1250: wave32, WMMA 16x16x32 bf16 (f32 accumulate).
// Fragment layouts per CDNA5 ISA:
//   A (16x32 bf16):  lane = {half=lane>>4, row=lane&15}; VGPR0..3 = K[8h..8h+7],
//                    VGPR4..7 = K[16+8h..16+8h+7]  -> two contiguous 16B chunks.
//   B (32x16):       symmetric; lane indexes the N column, same K striping,
//                    i.e. load from a row-major [N][K] (= B^T) array.
//   C/D (16x16 f32): VGPR r: lanes 0-15 -> M=r, lanes 16-31 -> M=r+8; N=lane&15.
// Heavy GEMMs use a 32x32 register tile per wave (2 A-frags x 2 B-frags ->
// 4 WMMAs per k-step, 1 fragment load per WMMA).
// ---------------------------------------------------------------------------

typedef __bf16 bf16;
typedef __attribute__((ext_vector_type(16))) __bf16 v16bf;
typedef __attribute__((ext_vector_type(8)))  float  v8f;

// float -> bf16, round-to-nearest-even
__device__ __forceinline__ bf16 f2bf(float f) {
  unsigned u = __float_as_uint(f);
  u += 0x7FFFu + ((u >> 16) & 1u);
  unsigned short s = (unsigned short)(u >> 16);
  return __builtin_bit_cast(__bf16, s);
}

// Load one 16x32 operand fragment for this lane.
// row_ptr = &Mat[row][k0] in a row-major [rows][K] bf16 array (K-contiguous).
__device__ __forceinline__ v16bf ld_frag(const bf16* row_ptr, int half) {
  union { v16bf v; uint4 q[2]; } u;
  u.q[0] = *(const uint4*)(row_ptr + half * 8);        // K = 8h .. 8h+7
  u.q[1] = *(const uint4*)(row_ptr + half * 8 + 16);   // K = 16+8h .. 16+8h+7
  return u.v;
}

__device__ __forceinline__ v8f wmma_bf16(v16bf a, v16bf b, v8f c) {
  return __builtin_amdgcn_wmma_f32_16x16x32_bf16(
      /*neg_a=*/false, a, /*neg_b=*/false, b,
      /*c_mod=*/(short)0, c, /*reuse_a=*/false, /*reuse_b=*/false);
}

// ---------------------------------------------------------------------------
// Problem constants
// ---------------------------------------------------------------------------
#define NB 16
#define NC 512
#define NH 64
#define NW 64
#define NN 4096   // H*W
#define NM 1024   // pooled positions
#define NCB 64
#define NCG 256

// ---------------------------------------------------------------------------
// Kernel 1: 2x2 average pool  x[B][512][64][64] -> xp[B][512][1024] (fp32)
// ---------------------------------------------------------------------------
__global__ void pool_kernel(const float* __restrict__ x, float* __restrict__ xp) {
  int idx = blockIdx.x * 256 + threadIdx.x;             // B*512*1024 threads
  int m  = idx & (NM - 1);
  int bc = idx >> 10;                                   // b*512 + c
  int my = m >> 5, mx = m & 31;
  const float* px = x + ((size_t)bc << 12);             // 64*64 plane
  int r0 = (my * 2) * NW + mx * 2;
  xp[idx] = 0.25f * (px[r0] + px[r0 + 1] + px[r0 + NW] + px[r0 + NW + 1]);
}

// ---------------------------------------------------------------------------
// Kernel 2: w_out -> bf16
// ---------------------------------------------------------------------------
__global__ void cvt_kernel(const float* __restrict__ w, bf16* __restrict__ wb, int n) {
  int i = blockIdx.x * 256 + threadIdx.x;
  if (i < n) wb[i] = f2bf(w[i]);
}

// ---------------------------------------------------------------------------
// Kernel 3: conv1x1 GEMM (fp32 in, bf16 out).
//   Y[p][o] = sum_c X[c][p] * W[o][c] + bias[o]
//   X: [Cin][Np] fp32 (per batch), W: [Co][Cin] fp32.
//   STORE_T=false -> Y position-major [Np][Co]; true -> channel-major [Co][Np].
// Block: 256 thr (8 waves), tile 64 pos x 64 cout, K-step 32 via LDS (transpose
// + fp32->bf16 conversion happens on the LDS staging path).
// ---------------------------------------------------------------------------
template <bool STORE_T>
__global__ void conv_gemm_kernel(const float* __restrict__ Xall,
                                 const float* __restrict__ W,
                                 const float* __restrict__ bias,
                                 bf16* __restrict__ Yall,
                                 int Cin, int Np, int Co) {
  __shared__ bf16 lx[64 * 40];   // [p][k], ld=40 (rows stay 16B aligned)
  __shared__ bf16 lw[64 * 40];   // [o][k]

  int b = blockIdx.z;
  const float* X = Xall + (size_t)b * Cin * Np;
  bf16* Y = Yall + (size_t)b * (size_t)Np * Co;

  int p0 = blockIdx.x * 64;
  int o0 = blockIdx.y * 64;
  int t = threadIdx.x;
  int wv = t >> 5, lane = t & 31, half = lane >> 4, row = lane & 15;
  int ptile = wv & 3;        // 4 position tiles of 16
  int ostrip = wv >> 2;      // 2 cout strips of 32

  v8f acc0 = {}; v8f acc1 = {};

  for (int k0 = 0; k0 < Cin; k0 += 32) {
    __syncthreads();
    // X tile transposed: lx[p][k] = X[k0+k][p0+p]   (coalesced along p)
    #pragma unroll
    for (int i = 0; i < 8; ++i) {
      int idx = t + i * 256;                 // 64*32 = 2048 elements
      int p = idx & 63, k = idx >> 6;
      lx[p * 40 + k] = f2bf(X[(size_t)(k0 + k) * Np + p0 + p]);
    }
    // W tile: lw[o][k] = W[o0+o][k0+k]      (coalesced along k)
    #pragma unroll
    for (int i = 0; i < 8; ++i) {
      int idx = t + i * 256;
      int k = idx & 31, o = idx >> 5;
      lw[o * 40 + k] = f2bf(W[(size_t)(o0 + o) * Cin + k0 + k]);
    }
    __syncthreads();

    v16bf a  = ld_frag(&lx[(ptile * 16 + row) * 40], half);
    v16bf b0 = ld_frag(&lw[(ostrip * 32 + row) * 40], half);
    v16bf b1 = ld_frag(&lw[(ostrip * 32 + 16 + row) * 40], half);
    acc0 = wmma_bf16(a, b0, acc0);
    acc1 = wmma_bf16(a, b1, acc1);
  }

  int oa = o0 + ostrip * 32 + row;
  int ob = oa + 16;
  float ba = bias[oa], bb = bias[ob];
  #pragma unroll
  for (int r = 0; r < 8; ++r) {
    int pp = p0 + ptile * 16 + r + half * 8;
    if (STORE_T) {
      Y[(size_t)oa * Np + pp] = f2bf(acc0[r] + ba);
      Y[(size_t)ob * Np + pp] = f2bf(acc1[r] + bb);
    } else {
      Y[(size_t)pp * Co + oa] = f2bf(acc0[r] + ba);
      Y[(size_t)pp * Co + ob] = f2bf(acc1[r] + bb);
    }
  }
}

// ---------------------------------------------------------------------------
// Kernel 4: logits (per batch)
//   L[n][m] = sum_k theta[n][k] * phi[m][k]   (theta [4096][64], phi [1024][64])
// Block tile 128n x 64m; per wave 32n x 32m (2x2 frags, 4 WMMAs per k-step).
// Fragments loaded straight from global (operands are L2-resident).
// ---------------------------------------------------------------------------
__global__ void logits_kernel(const bf16* __restrict__ theta,
                              const bf16* __restrict__ phi,
                              float* __restrict__ L) {
  int n0 = blockIdx.x * 128;
  int m0 = blockIdx.y * 64;
  int t = threadIdx.x;
  int wv = t >> 5, lane = t & 31, half = lane >> 4, row = lane & 15;
  int ntile = wv & 3;        // 4 n-tiles of 32
  int mstrip = wv >> 2;      // 2 m-strips of 32

  const bf16* pa0 = theta + (size_t)(n0 + ntile * 32 + row) * NCB;
  const bf16* pa1 = pa0 + 16 * NCB;
  const bf16* pb0 = phi + (size_t)(m0 + mstrip * 32 + row) * NCB;
  const bf16* pb1 = pb0 + 16 * NCB;

  v8f acc00 = {}, acc01 = {}, acc10 = {}, acc11 = {};
  #pragma unroll
  for (int k0 = 0; k0 < NCB; k0 += 32) {
    v16bf a0 = ld_frag(pa0 + k0, half);
    v16bf a1 = ld_frag(pa1 + k0, half);
    v16bf b0 = ld_frag(pb0 + k0, half);
    v16bf b1 = ld_frag(pb1 + k0, half);
    acc00 = wmma_bf16(a0, b0, acc00);
    acc01 = wmma_bf16(a0, b1, acc01);
    acc10 = wmma_bf16(a1, b0, acc10);
    acc11 = wmma_bf16(a1, b1, acc11);
  }
  int mA = m0 + mstrip * 32 + row;
  #pragma unroll
  for (int r = 0; r < 8; ++r) {
    int nA = n0 + ntile * 32 + r + half * 8;
    L[(size_t)nA * NM + mA]             = acc00[r];
    L[(size_t)nA * NM + mA + 16]        = acc01[r];
    L[(size_t)(nA + 16) * NM + mA]      = acc10[r];
    L[(size_t)(nA + 16) * NM + mA + 16] = acc11[r];
  }
}

// ---------------------------------------------------------------------------
// Kernel 5: row softmax over M=1024 (per batch), fp32 in -> bf16 out.
// One wave32 per row; 32 elements per lane kept in registers.
// ---------------------------------------------------------------------------
__global__ void softmax_kernel(const float* __restrict__ L, bf16* __restrict__ A) {
  int row = blockIdx.x * 8 + (threadIdx.x >> 5);
  int lane = threadIdx.x & 31;
  const float* p = L + (size_t)row * NM;

  float v[32];
  float mx = -3.0e38f;
  #pragma unroll
  for (int j = 0; j < 32; ++j) { v[j] = p[j * 32 + lane]; mx = fmaxf(mx, v[j]); }
  #pragma unroll
  for (int s = 16; s > 0; s >>= 1) mx = fmaxf(mx, __shfl_xor(mx, s, 32));
  float sum = 0.0f;
  #pragma unroll
  for (int j = 0; j < 32; ++j) { v[j] = __expf(v[j] - mx); sum += v[j]; }
  #pragma unroll
  for (int s = 16; s > 0; s >>= 1) sum += __shfl_xor(sum, s, 32);
  float inv = 1.0f / sum;
  bf16* q = A + (size_t)row * NM;
  #pragma unroll
  for (int j = 0; j < 32; ++j) q[j * 32 + lane] = f2bf(v[j] * inv);
}

// ---------------------------------------------------------------------------
// Kernel 6: attention output (per batch) — the dominant GEMM (K = 1024).
//   O[n][c] = sum_m attn[n][m] * g[c][m]   (attn [4096][1024], g [256][1024])
// Block tile 128n x 64c; per wave 32n x 32c (2x2 frags, 4 WMMAs per k-step).
// ---------------------------------------------------------------------------
__global__ void attnout_kernel(const bf16* __restrict__ A,
                               const bf16* __restrict__ G,
                               bf16* __restrict__ O) {
  int n0 = blockIdx.x * 128;
  int c0 = blockIdx.y * 64;
  int t = threadIdx.x;
  int wv = t >> 5, lane = t & 31, half = lane >> 4, row = lane & 15;
  int ntile = wv & 3;        // 4 n-tiles of 32
  int cstrip = wv >> 2;      // 2 c-strips of 32

  const bf16* pa0 = A + (size_t)(n0 + ntile * 32 + row) * NM;
  const bf16* pa1 = pa0 + 16 * NM;
  const bf16* pb0 = G + (size_t)(c0 + cstrip * 32 + row) * NM;
  const bf16* pb1 = pb0 + 16 * NM;

  v8f acc00 = {}, acc01 = {}, acc10 = {}, acc11 = {};
  #pragma unroll 2
  for (int k0 = 0; k0 < NM; k0 += 32) {
    v16bf a0 = ld_frag(pa0 + k0, half);
    v16bf a1 = ld_frag(pa1 + k0, half);
    v16bf b0 = ld_frag(pb0 + k0, half);
    v16bf b1 = ld_frag(pb1 + k0, half);
    acc00 = wmma_bf16(a0, b0, acc00);
    acc01 = wmma_bf16(a0, b1, acc01);
    acc10 = wmma_bf16(a1, b0, acc10);
    acc11 = wmma_bf16(a1, b1, acc11);
  }
  int cA = c0 + cstrip * 32 + row;
  #pragma unroll
  for (int r = 0; r < 8; ++r) {
    int nA = n0 + ntile * 32 + r + half * 8;
    O[(size_t)nA * NCG + cA]             = f2bf(acc00[r]);
    O[(size_t)nA * NCG + cA + 16]        = f2bf(acc01[r]);
    O[(size_t)(nA + 16) * NCG + cA]      = f2bf(acc10[r]);
    O[(size_t)(nA + 16) * NCG + cA + 16] = f2bf(acc11[r]);
  }
}

// ---------------------------------------------------------------------------
// Kernel 7: output conv + residual (per batch)
//   y[o][n] = gamma * (sum_c Wb[o][c] * O[n][c] + b_out[o]) + x[o][n]
// Block tile 64o x 128n; per wave 32o x 32n (2x2 frags, 4 WMMAs per k-step).
// ---------------------------------------------------------------------------
__global__ void final_kernel(const bf16* __restrict__ Wb,
                             const bf16* __restrict__ O,
                             const float* __restrict__ b_out,
                             const float* __restrict__ x,
                             const float* __restrict__ gamma,
                             float* __restrict__ y) {
  int n0 = blockIdx.x * 128;
  int o0 = blockIdx.y * 64;
  int t = threadIdx.x;
  int wv = t >> 5, lane = t & 31, half = lane >> 4, row = lane & 15;
  int ostrip = wv & 1;       // 2 o-strips of 32
  int ntile = wv >> 1;       // 4 n-tiles of 32

  const bf16* pa0 = Wb + (size_t)(o0 + ostrip * 32 + row) * NCG;
  const bf16* pa1 = pa0 + 16 * NCG;
  const bf16* pb0 = O + (size_t)(n0 + ntile * 32 + row) * NCG;
  const bf16* pb1 = pb0 + 16 * NCG;

  v8f acc00 = {}, acc01 = {}, acc10 = {}, acc11 = {};
  #pragma unroll 2
  for (int k0 = 0; k0 < NCG; k0 += 32) {
    v16bf a0 = ld_frag(pa0 + k0, half);
    v16bf a1 = ld_frag(pa1 + k0, half);
    v16bf b0 = ld_frag(pb0 + k0, half);
    v16bf b1 = ld_frag(pb1 + k0, half);
    acc00 = wmma_bf16(a0, b0, acc00);
    acc01 = wmma_bf16(a0, b1, acc01);
    acc10 = wmma_bf16(a1, b0, acc10);
    acc11 = wmma_bf16(a1, b1, acc11);
  }
  float gm = gamma[0];
  int nA = n0 + ntile * 32 + row;
  #pragma unroll
  for (int r = 0; r < 8; ++r) {
    int oA = o0 + ostrip * 32 + r + half * 8;
    int oB = oA + 16;
    float ba = b_out[oA], bb = b_out[oB];
    size_t i00 = (size_t)oA * NN + nA;
    size_t i10 = (size_t)oB * NN + nA;
    y[i00]      = gm * (acc00[r] + ba) + x[i00];
    y[i00 + 16] = gm * (acc01[r] + ba) + x[i00 + 16];
    y[i10]      = gm * (acc10[r] + bb) + x[i10];
    y[i10 + 16] = gm * (acc11[r] + bb) + x[i10 + 16];
  }
}

// ---------------------------------------------------------------------------
// Host launcher
// ---------------------------------------------------------------------------
extern "C" void kernel_launch(void* const* d_in, const int* in_sizes, int n_in,
                              void* d_out, int out_size, void* d_ws, size_t ws_size,
                              hipStream_t stream) {
  const float* x       = (const float*)d_in[0];
  const float* w_theta = (const float*)d_in[1];
  const float* b_theta = (const float*)d_in[2];
  const float* w_phi   = (const float*)d_in[3];
  const float* b_phi   = (const float*)d_in[4];
  const float* w_g     = (const float*)d_in[5];
  const float* b_g     = (const float*)d_in[6];
  const float* w_out   = (const float*)d_in[7];
  const float* b_out   = (const float*)d_in[8];
  const float* gamma   = (const float*)d_in[9];
  float* y = (float*)d_out;

  // Workspace carving (~80 MB total)
  char* ws = (char*)d_ws;
  float* xp    = (float*)ws;  ws += (size_t)NB * NC * NM * 4;        // 33.5 MB
  bf16*  theta = (bf16*)ws;   ws += (size_t)NB * NN * NCB * 2;       //  8.4 MB
  bf16*  phi   = (bf16*)ws;   ws += (size_t)NB * NM * NCB * 2;       //  2.1 MB
  bf16*  g     = (bf16*)ws;   ws += (size_t)NB * NCG * NM * 2;       //  8.4 MB
  bf16*  wob   = (bf16*)ws;   ws += (size_t)NC * NCG * 2;            //  0.3 MB
  float* logit = (float*)ws;  ws += (size_t)NN * NM * 4;             // 16.8 MB (per-batch, reused)
  bf16*  attn  = (bf16*)ws;   ws += (size_t)NN * NM * 2;             //  8.4 MB (per-batch, reused)
  bf16*  onc   = (bf16*)ws;                                          //  2.1 MB (per-batch, reused)

  // Stage 0: pooled x (conv1x1 and avg-pool commute) + w_out conversion
  pool_kernel<<<(NB * NC * NM) / 256, 256, 0, stream>>>(x, xp);
  cvt_kernel<<<(NC * NCG + 255) / 256, 256, 0, stream>>>(w_out, wob, NC * NCG);

  // Stage 1: the three projection GEMMs (all batches)
  conv_gemm_kernel<false><<<dim3(NN / 64, 1, NB), 256, 0, stream>>>(
      x, w_theta, b_theta, theta, NC, NN, NCB);                      // theta [n][k]
  conv_gemm_kernel<false><<<dim3(NM / 64, 1, NB), 256, 0, stream>>>(
      xp, w_phi, b_phi, phi, NC, NM, NCB);                           // phi [m][k]
  conv_gemm_kernel<true><<<dim3(NM / 64, NCG / 64, NB), 256, 0, stream>>>(
      xp, w_g, b_g, g, NC, NM, NCG);                                 // g [c][m]

  // Stage 2: per-batch attention (stream-ordered => scratch reuse is safe)
  for (int b = 0; b < NB; ++b) {
    logits_kernel<<<dim3(NN / 128, NM / 64), 256, 0, stream>>>(
        theta + (size_t)b * NN * NCB, phi + (size_t)b * NM * NCB, logit);
    softmax_kernel<<<NN / 8, 256, 0, stream>>>(logit, attn);
    attnout_kernel<<<dim3(NN / 128, NCG / 64), 256, 0, stream>>>(
        attn, g + (size_t)b * NCG * NM, onc);
    final_kernel<<<dim3(NN / 128, NC / 64), 256, 0, stream>>>(
        wob, onc, b_out, x + (size_t)b * NC * NN, gamma, y + (size_t)b * NC * NN);
  }
}
